// CrossAttention_17952963297653
// MI455X (gfx1250) — matrix operand order repaired and checked by reading, not compiled, not run on hardware
//
#include <hip/hip_runtime.h>

// ---------------------------------------------------------------------------
// CDNA5 (gfx1250) fused cross-attention.
//  - All matmuls: v_wmma_f32_16x16x32_bf16
//  - Attention K/V tiles double-buffered in LDS via global_load_async_to_lds
//    (ASYNCcnt), shared by the 4 waves of a block.
//  - Streaming tensors (bias / mask / logits) use non-temporal hints so the
//    ~25MB K/V/Q working set stays resident in the 192MB L2.
// ---------------------------------------------------------------------------

typedef __attribute__((ext_vector_type(16))) __bf16 v16bf;
typedef __attribute__((ext_vector_type(8)))  float  v8f;
typedef unsigned short u16b;

#define NB   4
#define NH   12
#define NQ   1024
#define NL   2048
#define CQd  768
#define CKV  512
#define HD   64
#define MASK_FILL (-65504.0f)

// LDS staging geometry (rows padded to 16B-aligned strides to spread banks)
#define K_LD   72            // 64 + 8 pad (elements)
#define V_LD   40            // 32 + 8 pad (elements)
#define K_TE   (32 * K_LD)   // elements per K tile buffer
#define V_TE   (64 * V_LD)   // elements per V tile buffer

__device__ __forceinline__ u16b f2bf(float f) {
  unsigned int u = __float_as_uint(f);
  return (u16b)((u + 0x7FFFu + ((u >> 16) & 1u)) >> 16);  // RNE f32->bf16
}

union FragU { uint4 q[2]; v16bf v; };

// A-matrix 16x32 bf16 fragment (ISA 7.12.2): lanes 0-15: M=lane, K=0-7/16-23;
// lanes 16-31: K=8-15/24-31.
__device__ __forceinline__ v16bf load_fragA(const u16b* src, int ld) {
  int lane = threadIdx.x & 31;
  const u16b* p = src + (long long)(lane & 15) * ld + ((lane >> 4) << 3);
  FragU f;
  f.q[0] = *(const uint4*)(p);
  f.q[1] = *(const uint4*)(p + 16);
  return f.v;
}

// B-matrix 32x16 bf16 fragment: lane = column N; lanes 0-15: K=0-7/16-23,
// lanes 16-31: K=8-15/24-31 (same K order as A).
__device__ __forceinline__ v16bf load_fragB(const u16b* src, int ld) {
  int lane = threadIdx.x & 31;
  const u16b* p = src + (long long)(lane & 15) * ld + ((lane >> 4) << 3);
  FragU f;
  f.q[0] = *(const uint4*)(p);
  f.q[1] = *(const uint4*)(p + 16);
  return f.v;
}

__device__ __forceinline__ v8f wmma_bf16(v16bf a, v16bf b, v8f c) {
  return __builtin_amdgcn_wmma_f32_16x16x32_bf16(false, a, false, b,
                                                 (short)0, c, false, false);
}

// Async global -> LDS copy, 16B per lane.  LDS offset = low 32 bits of the
// generic address (ISA 10.2: LDS aperture addr[31:0] is the LDS byte address).
__device__ __forceinline__ unsigned lds_off(const void* p) {
  return (unsigned)(unsigned long long)p;
}
__device__ __forceinline__ void async_g2l_b128(unsigned ldsoff, const void* gsrc) {
  asm volatile("global_load_async_to_lds_b128 %0, %1, off"
               :: "v"(ldsoff), "v"(gsrc) : "memory");
}
__device__ __forceinline__ void wait_async0() {
  asm volatile("s_wait_asynccnt 0" ::: "memory");
}

// ---------------------------------------------------------------------------
__global__ __launch_bounds__(256)
void f32_to_bf16_kernel(const float* __restrict__ in, u16b* __restrict__ out, int n) {
  for (int i = blockIdx.x * blockDim.x + threadIdx.x; i < n; i += gridDim.x * blockDim.x) {
    out[i] = f2bf(__builtin_nontemporal_load(&in[i]));
  }
}

// ---------------------------------------------------------------------------
// Generic C = A * W^T (both row-major bf16).  Wave tile 16x64, block 64x64.
__global__ __launch_bounds__(128)
void gemm_wmma_kernel(const u16b* __restrict__ A, int lda, long long sA,
                      const u16b* __restrict__ W, int ldb, long long sW,
                      u16b* __restrict__ outb, float* __restrict__ outf,
                      long long sC, int ldc, int K,
                      const float* __restrict__ bias) {
  int z = blockIdx.z;
  const u16b* Ab = A + (long long)z * sA;
  const u16b* Wb = W + (long long)z * sW;
  int wave = threadIdx.x >> 5;
  int m0 = blockIdx.x * 64 + wave * 16;
  int n0 = blockIdx.y * 64;

  v8f acc[4] = {};
  for (int k = 0; k < K; k += 32) {
    // issue all loads first so WMMAs can retire on partial loadcnt waits
    v16bf a  = load_fragA(Ab + (long long)m0 * lda + k, lda);
    v16bf b0 = load_fragB(Wb + (long long)(n0 +  0) * ldb + k, ldb);
    v16bf b1 = load_fragB(Wb + (long long)(n0 + 16) * ldb + k, ldb);
    v16bf b2 = load_fragB(Wb + (long long)(n0 + 32) * ldb + k, ldb);
    v16bf b3 = load_fragB(Wb + (long long)(n0 + 48) * ldb + k, ldb);
    acc[0] = wmma_bf16(a, b0, acc[0]);
    acc[1] = wmma_bf16(a, b1, acc[1]);
    acc[2] = wmma_bf16(a, b2, acc[2]);
    acc[3] = wmma_bf16(a, b3, acc[3]);
  }

  int lane = threadIdx.x & 31;
  int col  = lane & 15;
  int rb   = (lane >> 4) * 8;   // C-layout: VGPR i -> row rb+i
  if (outf) {
    float* o = outf + (long long)z * sC;
#pragma unroll
    for (int j = 0; j < 4; j++) {
      int gc = n0 + j * 16 + col;
      float bv = bias ? bias[gc] : 0.0f;
#pragma unroll
      for (int i = 0; i < 8; i++)
        __builtin_nontemporal_store(acc[j][i] + bv,
                                    &o[(long long)(m0 + rb + i) * ldc + gc]);
    }
  } else {
    u16b* o = outb + (long long)z * sC;
#pragma unroll
    for (int j = 0; j < 4; j++) {
      int gc = n0 + j * 16 + col;
#pragma unroll
      for (int i = 0; i < 8; i++)
        o[(long long)(m0 + rb + i) * ldc + gc] = f2bf(acc[j][i]);
    }
  }
}

// ---------------------------------------------------------------------------
// Fused attention.  Block = 4 waves sharing one (b,h); each wave owns 16 query
// rows.  K/V tiles (32 keys) double-buffered in LDS via async copies.
__global__ __launch_bounds__(128)
void attn_kernel(const u16b* __restrict__ qp,    // [B*NQ, CQ] bf16
                 const u16b* __restrict__ kp,    // [B*NL, CQ] bf16
                 const u16b* __restrict__ vtp,   // [B, CQ, NL] bf16 (V^T)
                 const float* __restrict__ bias, // [NH, NQ, NL]
                 const unsigned char* __restrict__ mask, // [B, NQ, NL]
                 float* __restrict__ attn_out,   // [B, NH, NQ, NL]
                 u16b* __restrict__ xheads) {    // [B*NQ, CQ] bf16
  __shared__ u16b Kst[2][K_TE];       // 32 keys x 64 d (padded rows)
  __shared__ u16b Vst[2][V_TE];       // 64 d   x 32 keys (padded rows)
  __shared__ u16b pst[4][16 * 32];    // per-wave P restage tile

  int t    = threadIdx.x;             // 0..127
  int wave = t >> 5;
  int lane = t & 31;
  int half = lane >> 4;
  int col  = lane & 15;
  int bh = blockIdx.y, b = bh / NH, h = bh % NH;
  int n0 = blockIdx.x * 64 + wave * 16;
  const float scale = 0.125f;         // 1/sqrt(64)

  const u16b* kbase = kp + ((long long)b * NL) * CQd + h * HD;        // keys
  const u16b* vbase = vtp + ((long long)b * CQd + h * HD) * NL;       // V^T
  const float* biasrow = bias + ((long long)h * NQ + n0) * NL;
  const unsigned char* maskrow = mask + ((long long)(b * NQ + n0)) * NL;
  float* attnrow = attn_out + (((long long)(b * NH + h)) * NQ + n0) * NL;

  // cooperative async tile fetch: K tile = 32x128B, V tile = 64x64B
  auto issue_tiles = [&](int buf, int l0) {
#pragma unroll
    for (int i = 0; i < 2; i++) {            // K: 256 x 16B chunks
      int c = t + i * 128;
      int row = c >> 3, co = (c & 7) * 8;
      async_g2l_b128(lds_off(&Kst[buf][row * K_LD + co]),
                     kbase + (long long)(l0 + row) * CQd + co);
    }
#pragma unroll
    for (int i = 0; i < 2; i++) {            // V: 256 x 16B chunks
      int c = t + i * 128;
      int row = c >> 2, co = (c & 3) * 8;
      async_g2l_b128(lds_off(&Vst[buf][row * V_LD + co]),
                     vbase + (long long)row * NL + l0 + co);
    }
  };

  const u16b* qrow = qp + ((long long)(b * NQ + n0)) * CQd + h * HD;
  v16bf qa0 = load_fragA(qrow,      CQd);    // d 0..31
  v16bf qa1 = load_fragA(qrow + 32, CQd);    // d 32..63

  v8f o[4] = {};
  float mrow[8], lrow[8];
#pragma unroll
  for (int i = 0; i < 8; i++) { mrow[i] = -3.0e38f; lrow[i] = 0.0f; }

  issue_tiles(0, 0);                          // prologue fill

  for (int it = 0; it < NL / 32; it++) {
    int l0  = it * 32;
    int buf = it & 1;
    wait_async0();                 // our copies into `buf` are complete
    __syncthreads();               // everyone's are; prev readers done
    if (it + 1 < NL / 32) issue_tiles(buf ^ 1, l0 + 32);  // overlap next fetch

    // prefetch next bias line (streaming, NT)
    __builtin_prefetch(&biasrow[(long long)(half * 8) * NL + l0 + 32 + col], 0, 0);

    // ---- S = Q K^T from LDS K tile ----
    v16bf bk00 = load_fragB(&Kst[buf][ 0 * K_LD +  0], K_LD);
    v16bf bk01 = load_fragB(&Kst[buf][ 0 * K_LD + 32], K_LD);
    v16bf bk10 = load_fragB(&Kst[buf][16 * K_LD +  0], K_LD);
    v16bf bk11 = load_fragB(&Kst[buf][16 * K_LD + 32], K_LD);
    v8f s0 = {}, s1 = {};
    s0 = wmma_bf16(qa0, bk00, s0);
    s0 = wmma_bf16(qa1, bk01, s0);
    s1 = wmma_bf16(qa0, bk10, s1);
    s1 = wmma_bf16(qa1, bk11, s1);

    // ---- scale, mask, +bias; single NT write of raw logits ----
    float p0[8], p1[8];
#pragma unroll
    for (int i = 0; i < 8; i++) {
      int row = half * 8 + i;
      long long off = (long long)row * NL + l0 + col;
      unsigned char m0c = __builtin_nontemporal_load(&maskrow[off]);
      unsigned char m1c = __builtin_nontemporal_load(&maskrow[off + 16]);
      float v0 = m0c ? MASK_FILL : s0[i] * scale;
      float v1 = m1c ? MASK_FILL : s1[i] * scale;
      v0 += __builtin_nontemporal_load(&biasrow[off]);
      v1 += __builtin_nontemporal_load(&biasrow[off + 16]);
      __builtin_nontemporal_store(v0, &attnrow[off]);
      __builtin_nontemporal_store(v1, &attnrow[off + 16]);
      p0[i] = v0; p1[i] = v1;
    }
    // ---- online softmax: row reductions across the 16-lane half ----
#pragma unroll
    for (int i = 0; i < 8; i++) {
      float tmx = fmaxf(p0[i], p1[i]);
      tmx = fmaxf(tmx, __shfl_xor(tmx, 1)); tmx = fmaxf(tmx, __shfl_xor(tmx, 2));
      tmx = fmaxf(tmx, __shfl_xor(tmx, 4)); tmx = fmaxf(tmx, __shfl_xor(tmx, 8));
      float newm = fmaxf(mrow[i], tmx);
      float fsc = __expf(mrow[i] - newm);
      mrow[i] = newm;
      p0[i] = __expf(p0[i] - newm);
      p1[i] = __expf(p1[i] - newm);
      float s = p0[i] + p1[i];
      s += __shfl_xor(s, 1); s += __shfl_xor(s, 2);
      s += __shfl_xor(s, 4); s += __shfl_xor(s, 8);
      lrow[i] = lrow[i] * fsc + s;
#pragma unroll
      for (int j = 0; j < 4; j++) o[j][i] *= fsc;
    }
    // ---- restage P (C-layout -> A-layout) in this wave's LDS tile ----
#pragma unroll
    for (int i = 0; i < 8; i++) {
      int row = half * 8 + i;
      pst[wave][row * 32 + col]      = f2bf(p0[i]);
      pst[wave][row * 32 + 16 + col] = f2bf(p1[i]);
    }
    __builtin_amdgcn_wave_barrier();   // DS ops are in-order per wave
    v16bf pa = load_fragA(pst[wave], 32);

    // ---- O += P * V from LDS V^T tile ----
    v16bf vb0 = load_fragB(&Vst[buf][ 0 * V_LD], V_LD);
    v16bf vb1 = load_fragB(&Vst[buf][16 * V_LD], V_LD);
    v16bf vb2 = load_fragB(&Vst[buf][32 * V_LD], V_LD);
    v16bf vb3 = load_fragB(&Vst[buf][48 * V_LD], V_LD);
    o[0] = wmma_bf16(pa, vb0, o[0]);
    o[1] = wmma_bf16(pa, vb1, o[1]);
    o[2] = wmma_bf16(pa, vb2, o[2]);
    o[3] = wmma_bf16(pa, vb3, o[3]);
  }

  // ---- normalize and emit per-head output (bf16) ----
#pragma unroll
  for (int i = 0; i < 8; i++) {
    float inv = 1.0f / lrow[i];
#pragma unroll
    for (int j = 0; j < 4; j++) o[j][i] *= inv;
  }
  u16b* xrow = xheads + ((long long)(b * NQ + n0)) * CQd + h * HD;
#pragma unroll
  for (int j = 0; j < 4; j++)
#pragma unroll
    for (int i = 0; i < 8; i++)
      xrow[(long long)(half * 8 + i) * CQd + j * 16 + col] = f2bf(o[j][i]);
}

// ---------------------------------------------------------------------------
extern "C" void kernel_launch(void* const* d_in, const int* in_sizes, int n_in,
                              void* d_out, int out_size, void* d_ws, size_t ws_size,
                              hipStream_t stream) {
  const float* query  = (const float*)d_in[0];
  const float* memory = (const float*)d_in[1];
  const float* bias   = (const float*)d_in[2];
  const unsigned char* mask = (const unsigned char*)d_in[3];
  const float* Wq     = (const float*)d_in[4];
  const float* Wkv    = (const float*)d_in[5];
  const float* Wproj  = (const float*)d_in[6];
  const float* bproj  = (const float*)d_in[7];

  float* x_out    = (float*)d_out;
  float* attn_out = x_out + (size_t)NB * NQ * CQd;

  char* ws = (char*)d_ws;
  size_t off = 0;
  auto carve = [&](size_t bytes) -> void* {
    void* p = ws + off; off += (bytes + 255) & ~(size_t)255; return p;
  };
  u16b* querybf = (u16b*)carve((size_t)NB * NQ * CQd * 2);
  u16b* membf   = (u16b*)carve((size_t)NB * NL * CKV * 2);
  u16b* wqbf    = (u16b*)carve((size_t)CQd * CQd * 2);
  u16b* wkvbf   = (u16b*)carve((size_t)2 * CQd * CKV * 2);
  u16b* wpbf    = (u16b*)carve((size_t)CQd * CQd * 2);
  u16b* qp      = (u16b*)carve((size_t)NB * NQ * CQd * 2);
  u16b* kp      = (u16b*)carve((size_t)NB * NL * CQd * 2);
  u16b* vtp     = (u16b*)carve((size_t)NB * CQd * NL * 2);
  u16b* xh      = (u16b*)carve((size_t)NB * NQ * CQd * 2);

  auto cvt = [&](const float* src, u16b* dst, int n) {
    int blocks = (n + 255) / 256; if (blocks > 4096) blocks = 4096;
    f32_to_bf16_kernel<<<blocks, 256, 0, stream>>>(src, dst, n);
  };
  cvt(query,  querybf, NB * NQ * CQd);
  cvt(memory, membf,   NB * NL * CKV);
  cvt(Wq,     wqbf,    CQd * CQd);
  cvt(Wkv,    wkvbf,   2 * CQd * CKV);
  cvt(Wproj,  wpbf,    CQd * CQd);

  // Q proj: [4096,768] = query * Wq^T
  gemm_wmma_kernel<<<dim3(64, 12, 1), 128, 0, stream>>>(
      querybf, CQd, 0, wqbf, CQd, 0, qp, nullptr, 0, CQd, CQd, nullptr);

  // K proj: [8192,768] = memory * Wk^T   (Wkv rows 0..767)
  gemm_wmma_kernel<<<dim3(128, 12, 1), 128, 0, stream>>>(
      membf, CKV, 0, wkvbf, CKV, 0, kp, nullptr, 0, CQd, CKV, nullptr);

  // V^T proj: per batch [768,2048] = Wv * memory^T  (Wkv rows 768..1535)
  gemm_wmma_kernel<<<dim3(12, 32, NB), 128, 0, stream>>>(
      wkvbf + (size_t)CQd * CKV, CKV, 0,
      membf, CKV, (long long)NL * CKV,
      vtp, nullptr, (long long)CQd * NL, NL, CKV, nullptr);

  // fused attention (async double-buffered K/V, online softmax, NT streaming)
  attn_kernel<<<dim3(NQ / 64, NB * NH), 128, 0, stream>>>(
      qp, kp, vtp, bias, mask, attn_out, xh);

  // out proj: x = xheads * Wproj^T + bproj  (f32 epilogue)
  gemm_wmma_kernel<<<dim3(64, 12, 1), 128, 0, stream>>>(
      xh, CQd, 0, wpbf, CQd, 0, nullptr, x_out, 0, CQd, CQd, bproj);
}
